// Switch_30923764531756
// MI455X (gfx1250) — matrix operand order repaired and checked by reading, not compiled, not run on hardware
//
#include <hip/hip_runtime.h>
#include <hip/hip_bf16.h>
#include <math.h>

// Problem dims (fixed by the reference)
#define B_ 8
#define S_ 512
#define D_ 1024
#define F_ 4096
#define E_ 8
#define T_ (B_*S_)      // 4096 tokens
#define C_ (T_/E_)      // 512 expert capacity

typedef __attribute__((ext_vector_type(16))) _Float16 hvec16;
typedef __attribute__((ext_vector_type(8)))  _Float16 hvec8;
typedef __attribute__((ext_vector_type(8)))  float    fvec8;

// ---------------------------------------------------------------- zero output
__global__ void zero4_kernel(float4* __restrict__ p, int n) {
  int i = blockIdx.x * blockDim.x + threadIdx.x;
  if (i < n) p[i] = make_float4(0.f, 0.f, 0.f, 0.f);
}

// ---------------------------------------------------------------- router
// One block (256 threads) per token: logits = x_t @ w_route + b_route,
// gate = max softmax prob = 1/sum(exp(l - lmax)), idx = argmax.
__global__ void router_kernel(const float* __restrict__ x,
                              const float* __restrict__ wr,
                              const float* __restrict__ br,
                              float* __restrict__ gate,
                              int* __restrict__ eidx) {
  __shared__ float red[256 * E_];
  const int t   = blockIdx.x;
  const int tid = threadIdx.x;
  const float* xt = x + (size_t)t * D_;
  float acc[E_];
#pragma unroll
  for (int e = 0; e < E_; e++) acc[e] = 0.f;
  const int d0 = tid * 4;                      // 256*4 == D
  const float4 xv = *(const float4*)(xt + d0);
  const float xa[4] = {xv.x, xv.y, xv.z, xv.w};
#pragma unroll
  for (int j = 0; j < 4; j++) {
    const float4 w0 = *(const float4*)(wr + (size_t)(d0 + j) * E_);
    const float4 w1 = *(const float4*)(wr + (size_t)(d0 + j) * E_ + 4);
    acc[0] += xa[j] * w0.x; acc[1] += xa[j] * w0.y;
    acc[2] += xa[j] * w0.z; acc[3] += xa[j] * w0.w;
    acc[4] += xa[j] * w1.x; acc[5] += xa[j] * w1.y;
    acc[6] += xa[j] * w1.z; acc[7] += xa[j] * w1.w;
  }
#pragma unroll
  for (int e = 0; e < E_; e++) red[tid * E_ + e] = acc[e];
  __syncthreads();
  for (int s = 128; s > 0; s >>= 1) {
    if (tid < s) {
#pragma unroll
      for (int e = 0; e < E_; e++) red[tid*E_+e] += red[(tid+s)*E_+e];
    }
    __syncthreads();
  }
  if (tid == 0) {
    float lg[E_]; float mx = -1e30f; int am = 0;
#pragma unroll
    for (int e = 0; e < E_; e++) {
      lg[e] = red[e] + br[e];
      if (lg[e] > mx) { mx = lg[e]; am = e; }
    }
    float sum = 0.f;
#pragma unroll
    for (int e = 0; e < E_; e++) sum += expf(lg[e] - mx);
    gate[t] = 1.f / sum;   // max softmax prob
    eidx[t] = am;
  }
}

// ---------------------------------------------------------------- assignment
// Single block, 256 threads x 16 tokens each. Hillis-Steele scan of per-expert
// counts gives each token its 1-based position within its expert (token order).
// pos >= C -> dropped (gate zeroed). token_of[e*C + pos] = t; slot 0 empty.
__global__ void assign_kernel(const int* __restrict__ eidx,
                              float* __restrict__ gate,
                              int* __restrict__ token_of) {
  __shared__ int cnt[2][256 * E_];
  const int tid = threadIdx.x;
  for (int i = tid; i < E_ * C_; i += 256) token_of[i] = -1;
  int loc[E_];
#pragma unroll
  for (int e = 0; e < E_; e++) loc[e] = 0;
  int myidx[16];
#pragma unroll
  for (int j = 0; j < 16; j++) {
    myidx[j] = eidx[tid * 16 + j];
    loc[myidx[j]]++;
  }
#pragma unroll
  for (int e = 0; e < E_; e++) cnt[0][tid*E_+e] = loc[e];
  __syncthreads();
  int src = 0;
  for (int s = 1; s < 256; s <<= 1) {
#pragma unroll
    for (int e = 0; e < E_; e++) {
      int v = cnt[src][tid*E_+e];
      if (tid >= s) v += cnt[src][(tid-s)*E_+e];
      cnt[1-src][tid*E_+e] = v;
    }
    __syncthreads();
    src = 1 - src;
  }
  int off[E_];
#pragma unroll
  for (int e = 0; e < E_; e++) off[e] = (tid > 0) ? cnt[src][(tid-1)*E_+e] : 0;
#pragma unroll
  for (int j = 0; j < 16; j++) {
    const int t = tid * 16 + j;
    const int e = myidx[j];
    const int pos = ++off[e];              // 1-based position
    if (pos < C_) token_of[e * C_ + pos] = t;
    else          gate[t] = 0.f;           // dropped token
  }
}

// ---------------------------------------------------------------- dispatch
// Xe[e,c,:] = gate_t * x_t  (f16), zeros for empty slots. One block per (e,c).
__global__ void dispatch_kernel(const float* __restrict__ x,
                                const float* __restrict__ gate,
                                const int* __restrict__ token_of,
                                _Float16* __restrict__ Xe) {
  const int ec  = blockIdx.x;
  const int tid = threadIdx.x;             // 128 threads * 8 elems = D
  _Float16* dst = Xe + (size_t)ec * D_ + tid * 8;
  const int t = token_of[ec];
  hvec8 h;
  if (t < 0) {
#pragma unroll
    for (int q = 0; q < 8; q++) h[q] = (_Float16)0.f;
  } else {
    const float g = gate[t];
    const float* src = x + (size_t)t * D_ + tid * 8;
    const float4 a = *(const float4*)(src);
    const float4 b = *(const float4*)(src + 4);
    h[0]=(_Float16)(g*a.x); h[1]=(_Float16)(g*a.y); h[2]=(_Float16)(g*a.z); h[3]=(_Float16)(g*a.w);
    h[4]=(_Float16)(g*b.x); h[5]=(_Float16)(g*b.y); h[6]=(_Float16)(g*b.z); h[7]=(_Float16)(g*b.w);
  }
  *(hvec8*)dst = h;
}

// ---------------------------------------------------------------- grouped GEMM
// C[M,N] = A[M,K](f16) * B[K,N](f32->f16) per expert, fp32 accumulation via
// v_wmma_f32_16x16x32_f16. 128x128 tile, BK=32, 4 waves each own 64x64.
// Double-buffered LDS: A tile filled by GLOBAL_LOAD_ASYNC_TO_LDS_B128 (ASYNCcnt),
// B tile converted fp32->f16 with K-pair packed ds_store_b32.
// MODE 0: out = gelu_exact(acc + b1) -> f16 H buffer.
// MODE 1: scatter out[token_of[e,m]] = gate * (acc + b2)  (fused combine).
#define BM 128
#define BN 128
#define BK 32
#define LDP (BK + 8)   // LDS row pitch in halves (80B: conflict-free frag reads)

template <int MODE>
__global__ __launch_bounds__(128)
void moe_gemm_kernel(const _Float16* __restrict__ A,
                     const float* __restrict__ Bw,
                     const float* __restrict__ bias,
                     void* __restrict__ Out,
                     const int* __restrict__ token_of,
                     const float* __restrict__ gate,
                     int M, int N, int K) {
  __shared__ _Float16 Al[2][BM * LDP];  // 2 x 10 KB
  __shared__ _Float16 Bl[2][BN * LDP];  // 2 x 10 KB, stored [n][k] (K-contig)
  const int e  = blockIdx.z;
  const int m0 = blockIdx.x * BM;       // m-tile fastest -> weight slab L2 reuse
  const int n0 = blockIdx.y * BN;
  const int tid  = threadIdx.x;
  const int lane = tid & 31;
  const int wave = tid >> 5;
  const int wr = (wave >> 1) * 64;
  const int wc = (wave & 1) * 64;
  const int l16 = lane & 15;
  const int lh  = lane >> 4;
  const _Float16* Ae = A  + (size_t)e * M * K;
  const float*    Be = Bw + (size_t)e * K * N;

  // Wave-relative LDS byte address = low 32 bits of the generic pointer
  const unsigned ldsA0 = (unsigned)(size_t)(void*)&Al[0][0];

  fvec8 acc[4][4] = {};

  // Fire-and-forget DMA of the (f16, K-contiguous) A tile into LDS.
  auto issue_asyncA = [&](int kk, int buf) {
#pragma unroll
    for (int p = 0; p < 4; p++) {
      const int r = p * 32 + (tid >> 2);
      const int c = (tid & 3) * 8;
      const _Float16* g = Ae + (size_t)(m0 + r) * K + kk + c;
      const unsigned lds = ldsA0 + (unsigned)((buf * BM * LDP + r * LDP + c) * 2);
      asm volatile("global_load_async_to_lds_b128 %0, %1, off"
                   :: "v"(lds), "v"(g) : "memory");
    }
  };

  // B tile: each thread reads 2 adjacent K rows x 4 cols (2x b128, coalesced),
  // packs K-pairs into dwords, stores transposed with ds_store_b32.
  auto loadB = [&](int kk, int buf) {
    _Float16* bb = &Bl[buf][0];
#pragma unroll
    for (int p = 0; p < 4; p++) {
      const int q  = p * 128 + tid;           // 0..511
      const int kr = (q >> 5) * 2;            // even k row
      const int cc = (q & 31) * 4;            // col group
      const float4 v0 = *(const float4*)(Be + (size_t)(kk + kr)     * N + n0 + cc);
      const float4 v1 = *(const float4*)(Be + (size_t)(kk + kr + 1) * N + n0 + cc);
      union { _Float16 h[2]; unsigned u; } pk;
      pk.h[0]=(_Float16)v0.x; pk.h[1]=(_Float16)v1.x; *(unsigned*)&bb[(cc+0)*LDP+kr]=pk.u;
      pk.h[0]=(_Float16)v0.y; pk.h[1]=(_Float16)v1.y; *(unsigned*)&bb[(cc+1)*LDP+kr]=pk.u;
      pk.h[0]=(_Float16)v0.z; pk.h[1]=(_Float16)v1.z; *(unsigned*)&bb[(cc+2)*LDP+kr]=pk.u;
      pk.h[0]=(_Float16)v0.w; pk.h[1]=(_Float16)v1.w; *(unsigned*)&bb[(cc+3)*LDP+kr]=pk.u;
    }
  };

  // Fragments per ISA 16-bit layouts:
  // A 16x32: lane halves hold K in [lh*8, lh*8+8) and [lh*8+16, lh*8+24)
  // B 32x16: lane holds column l16, K in [lh*8, lh*8+8) and [lh*8+16, lh*8+24)
  auto compute = [&](int buf) {
    hvec16 af[4], bf[4];
#pragma unroll
    for (int i = 0; i < 4; i++) {
      const int r = wr + i*16 + l16;
      const hvec8 lo = *(const hvec8*)&Al[buf][r*LDP + lh*8];
      const hvec8 hi = *(const hvec8*)&Al[buf][r*LDP + lh*8 + 16];
#pragma unroll
      for (int q = 0; q < 8; q++) { af[i][q] = lo[q]; af[i][q+8] = hi[q]; }
    }
#pragma unroll
    for (int j = 0; j < 4; j++) {
      const int c = wc + j*16 + l16;
      const hvec8 blo = *(const hvec8*)&Bl[buf][c*LDP + lh*8];
      const hvec8 bhi = *(const hvec8*)&Bl[buf][c*LDP + lh*8 + 16];
#pragma unroll
      for (int q = 0; q < 8; q++) { bf[j][q] = blo[q]; bf[j][q+8] = bhi[q]; }
    }
#pragma unroll
    for (int i = 0; i < 4; i++)
#pragma unroll
      for (int j = 0; j < 4; j++)
        acc[i][j] = __builtin_amdgcn_wmma_f32_16x16x32_f16(
            false, af[i], false, bf[j], (short)0, acc[i][j], false, false);
  };

  // Prologue: fill buffer 0.
  issue_asyncA(0, 0);
  loadB(0, 0);
  asm volatile("s_wait_asynccnt 0x0" ::: "memory");
  __syncthreads();

  int cur = 0;
  for (int kk = 0; kk < K; kk += BK) {
    const int nxt = cur ^ 1;
    const bool more = (kk + BK) < K;
    if (more) issue_asyncA(kk + BK, nxt);   // DMA overlaps the WMMAs below
    compute(cur);
    if (more) loadB(kk + BK, nxt);          // VALU cvt overlaps XDL drain
    asm volatile("s_wait_asynccnt 0x0" ::: "memory");
    __syncthreads();                         // also waits dscnt for B stores
    cur = nxt;
  }

  // Epilogue. C/D layout: VGPR p holds (m_local = p + 8*lh, n_local = l16).
#pragma unroll
  for (int i = 0; i < 4; i++) {
#pragma unroll
    for (int j = 0; j < 4; j++) {
      const int gn = n0 + wc + j*16 + l16;
      const float bv = bias[(size_t)e * N + gn];
#pragma unroll
      for (int p = 0; p < 8; p++) {
        const int gm = m0 + wr + i*16 + lh*8 + p;
        const float v = acc[i][j][p] + bv;
        if (MODE == 0) {
          const float g = 0.5f * v * (1.f + erff(v * 0.70710678118654752f));
          ((_Float16*)Out)[(size_t)e * M * N + (size_t)gm * N + gn] = (_Float16)g;
        } else {
          const int t = token_of[e * M + gm];   // M == C here
          if (t >= 0)
            ((float*)Out)[(size_t)t * N + gn] = gate[t] * v;
        }
      }
    }
  }
}

// ---------------------------------------------------------------- launch
extern "C" void kernel_launch(void* const* d_in, const int* in_sizes, int n_in,
                              void* d_out, int out_size, void* d_ws, size_t ws_size,
                              hipStream_t stream) {
  (void)in_sizes; (void)n_in; (void)out_size; (void)ws_size;
  const float* x  = (const float*)d_in[0];
  const float* wr = (const float*)d_in[1];
  const float* br = (const float*)d_in[2];
  const float* w1 = (const float*)d_in[3];
  const float* b1 = (const float*)d_in[4];
  const float* w2 = (const float*)d_in[5];
  const float* b2 = (const float*)d_in[6];
  float* out = (float*)d_out;

  // workspace carve-up (all 64KB-aligned slabs; ~40.2 MB total)
  char* ws = (char*)d_ws;
  float*    gate     = (float*)(ws + 0 * (64 << 10));           // T f32
  int*      eidx     = (int*)  (ws + 1 * (64 << 10));           // T i32
  int*      token_of = (int*)  (ws + 2 * (64 << 10));           // E*C i32
  _Float16* Xe       = (_Float16*)(ws + 3 * (64 << 10));        // E*C*D f16 (8 MB)
  _Float16* H        = (_Float16*)(ws + 3 * (64 << 10)
                                   + (size_t)E_ * C_ * D_ * 2); // E*C*F f16 (32 MB)

  zero4_kernel<<<(T_*D_/4 + 255)/256, 256, 0, stream>>>((float4*)out, T_*D_/4);
  router_kernel<<<T_, 256, 0, stream>>>(x, wr, br, gate, eidx);
  assign_kernel<<<1, 256, 0, stream>>>(eidx, gate, token_of);
  dispatch_kernel<<<E_*C_, 128, 0, stream>>>(x, gate, token_of, Xe);
  // GEMM1: [C x D] @ [D x F] + b1 -> gelu -> H (f16)
  moe_gemm_kernel<0><<<dim3(C_/BM, F_/BN, E_), 128, 0, stream>>>(
      Xe, w1, b1, (void*)H, nullptr, nullptr, C_, F_, D_);
  // GEMM2: [C x F] @ [F x D] + b2 -> gate-scaled scatter into out
  moe_gemm_kernel<1><<<dim3(C_/BM, D_/BN, E_), 128, 0, stream>>>(
      H, w2, b2, (void*)out, token_of, gate, C_, D_, F_);
}
